// MockRouter_76192719831328
// MI455X (gfx1250) — matrix operand order repaired and checked by reading, not compiled, not run on hardware
//
#include <hip/hip_runtime.h>
#include <hip/hip_bf16.h>
#include <math.h>

// ---------------------------------------------------------------------------
// MoE router on MI455X (gfx1250, wave32, WMMA).
//   logits = x[16384,4096](bf16) @ gate_w.T[4096,64](bf16)
//   scores = sigmoid(logits); top-2; normalize; out: bf16 weights + i32 idx.
//
// Roofline: 8.6 GFLOP vs 134 MB of x => memory-bound (~5.8us @ 23.3 TB/s).
// Saturating HBM needs ~12-14 MB in flight (BW x ~550ns). Design:
//   * 1024 waves (16 tokens/wave) ~= 2 per SIMD32 on a ~128-WGP part
//   * 3-deep rotating pipeline: stage s+2 loads issue while stage s computes,
//     all addressing affine (drain stages peeled) -> ~20 b128 loads in
//     flight per wave (~10 MB chip-wide) plus counter-free prefetches
//   * global_prefetch_b8 on the A stream for extra depth at zero LOADcnt cost
//   * gate_w (512 KB) stays L2-resident; x streamed from HBM exactly once.
// Both A and B use the same contiguous k order (half h: k = 16h..16h+15) from row-major
// memory (no transpose loads needed).
// ---------------------------------------------------------------------------

typedef __bf16 bf16_t;
typedef __attribute__((ext_vector_type(16))) __bf16 v16bf;
typedef __attribute__((ext_vector_type(8)))  __bf16 v8bf;   // one 16B load
typedef __attribute__((ext_vector_type(8)))  float  v8f;

#define DIM            4096
#define NEXPERTS       64
#define KSTEP          32
#define NITER          (DIM / KSTEP)          // 128
#define WAVE_TOKENS    16                     // 1 M-tile per wave
#define WAVES_PER_BLK  4
#define BLK_TOKENS     (WAVE_TOKENS * WAVES_PER_BLK)   // 64
#define PF_AHEAD       1024                   // A prefetch distance (bf16 elems)

union Pack { v16bf v; v8bf h[2]; };

// A-tile fragment: lane half h holds K = h*16..h*16+15.
__device__ __forceinline__ void load_a(Pack& p, const bf16_t* base, int k) {
    p.h[0] = *(const v8bf*)(base + k);
    p.h[1] = *(const v8bf*)(base + k + 8);
}
// B-tile fragment: lane half h holds K = h*16..h*16+15 (16 contiguous bf16
// from gate_w row n), split into two b128 loads.
__device__ __forceinline__ void load_b(Pack& p, const bf16_t* base, int k) {
    p.h[0] = *(const v8bf*)(base + k);
    p.h[1] = *(const v8bf*)(base + k + 8);
}

__global__ __launch_bounds__(WAVES_PER_BLK * 32, 1) void router_wmma_kernel(
    const bf16_t* __restrict__ x,      // [tokens, DIM] row-major
    const bf16_t* __restrict__ gw,     // [NEXPERTS, DIM] row-major
    bf16_t*       __restrict__ wout,   // [tokens, 2] bf16 weights
    int*          __restrict__ iout)   // [tokens, 2] int32 indices
{
    // +1 float pad kills stride-64 bank conflicts in the top-2 scan.
    __shared__ float logits[BLK_TOKENS][NEXPERTS + 1];   // ~16.6 KB

    const int lane = threadIdx.x & 31;
    const int wave = threadIdx.x >> 5;
    const int half = lane >> 4;        // 0: lanes 0-15, 1: lanes 16-31
    const int m    = lane & 15;
    const long tok0 = (long)blockIdx.x * BLK_TOKENS + (long)wave * WAVE_TOKENS;

    const bf16_t* ap  = x + (tok0 + m) * (long)DIM + half * 16;
    const bf16_t* b0p = gw + ( 0 + m) * DIM + half * 16;
    const bf16_t* b1p = gw + (16 + m) * DIM + half * 16;
    const bf16_t* b2p = gw + (32 + m) * DIM + half * 16;
    const bf16_t* b3p = gw + (48 + m) * DIM + half * 16;

    v8f acc0 = (v8f){0.f,0.f,0.f,0.f,0.f,0.f,0.f,0.f};
    v8f acc1 = acc0, acc2 = acc0, acc3 = acc0;

    // 3-deep rotating pipeline buffers.
    Pack A[3], B0[3], B1[3], B2[3], B3[3];

#define LOAD_STAGE(buf, k)                                                   \
    do {                                                                     \
        load_a(A[buf], ap, (k));                                             \
        load_b(B0[buf], b0p, (k));                                           \
        load_b(B1[buf], b1p, (k));                                           \
        load_b(B2[buf], b2p, (k));                                           \
        load_b(B3[buf], b3p, (k));                                           \
    } while (0)

#define WMMA_STAGE(buf)                                                      \
    do {                                                                     \
        acc0 = __builtin_amdgcn_wmma_f32_16x16x32_bf16(false, A[buf].v, false, B0[buf].v, (short)0, acc0, false, false); \
        acc1 = __builtin_amdgcn_wmma_f32_16x16x32_bf16(false, A[buf].v, false, B1[buf].v, (short)0, acc1, false, false); \
        acc2 = __builtin_amdgcn_wmma_f32_16x16x32_bf16(false, A[buf].v, false, B2[buf].v, (short)0, acc2, false, false); \
        acc3 = __builtin_amdgcn_wmma_f32_16x16x32_bf16(false, A[buf].v, false, B3[buf].v, (short)0, acc3, false, false); \
    } while (0)

    LOAD_STAGE(0, 0);
    LOAD_STAGE(1, KSTEP);

    // Main pipelined loop: all k offsets affine (no clamps), drain peeled.
    // Trip count NITER-2 = 126 is divisible by 3, so s%3 folds to constants.
#pragma unroll 3
    for (int s = 0; s < NITER - 2; ++s) {
        const int cur = s % 3;
        const int nx2 = (s + 2) % 3;
        LOAD_STAGE(nx2, (s + 2) * KSTEP);
        // Deep A prefetch: global_prefetch_b8, no LOADcnt, pure extra depth.
        __builtin_prefetch(ap + s * KSTEP + PF_AHEAD, 0, 3);
        WMMA_STAGE(cur);
    }
    WMMA_STAGE((NITER - 2) % 3);   // == 0
    WMMA_STAGE((NITER - 1) % 3);   // == 1

#undef LOAD_STAGE
#undef WMMA_STAGE

    // D-layout (32-bit C/D 16x16): VGPR r -> row M = r + half*8, col N = m.
    // Reference does the matmul in bf16 before .float(): round logits to bf16.
    {
        const int tl = wave * WAVE_TOKENS + half * 8;
        const v8f a[4] = {acc0, acc1, acc2, acc3};
#pragma unroll
        for (int t = 0; t < 4; ++t)
#pragma unroll
            for (int r = 0; r < 8; ++r)
                logits[tl + r][t * 16 + m] = (float)(__bf16)a[t][r];
    }

    __syncthreads();

    // One lane per token: top-2 over 64 experts (strict '>' keeps the
    // earliest index on ties, matching jax.lax.top_k; sigmoid is monotone
    // so ranking logits == ranking scores).
    const int tid = threadIdx.x;
    if (tid < BLK_TOKENS) {
        const long tok = (long)blockIdx.x * BLK_TOKENS + tid;
        float b1 = -INFINITY, b2 = -INFINITY;
        int   i1 = 0, i2 = 0;
#pragma unroll 8
        for (int e = 0; e < NEXPERTS; ++e) {
            const float v = logits[tid][e];
            if (v > b1)      { b2 = b1; i2 = i1; b1 = v; i1 = e; }
            else if (v > b2) { b2 = v;  i2 = e; }
        }
        const float s1 = 1.0f / (1.0f + expf(-b1));
        const float s2 = 1.0f / (1.0f + expf(-b2));
        float den = s1 + s2;
        den = den < 1e-12f ? 1e-12f : den;
        wout[2 * tok]     = (__bf16)(s1 / den);
        wout[2 * tok + 1] = (__bf16)(s2 / den);
        iout[2 * tok]     = i1;
        iout[2 * tok + 1] = i2;
    }
}

extern "C" void kernel_launch(void* const* d_in, const int* in_sizes, int n_in,
                              void* d_out, int out_size, void* d_ws, size_t ws_size,
                              hipStream_t stream) {
    const bf16_t* x  = (const bf16_t*)d_in[0];   // [tokens, 4096] bf16
    const bf16_t* gw = (const bf16_t*)d_in[1];   // [64, 4096] bf16

    const int tokens = in_sizes[0] / DIM;        // 16384

    // Tuple outputs concatenated flat in return order:
    //   [tokens*2] bf16 weights, then [tokens*2] int32 indices.
    bf16_t* wout = (bf16_t*)d_out;
    int*    iout = (int*)((char*)d_out + (size_t)tokens * 2 * sizeof(bf16_t));

    const int blocks = tokens / BLK_TOKENS;      // 256
    router_wmma_kernel<<<blocks, WAVES_PER_BLK * 32, 0, stream>>>(x, gw, wout, iout);
}
